// LSTM_VAE_45363444580936
// MI455X (gfx1250) — matrix operand order repaired and checked by reading, not compiled, not run on hardware
//
#include <hip/hip_runtime.h>
#include <cstddef>

// ---------------------------------------------------------------------------
// LSTM-VAE for MI455X (gfx1250): all GEMMs via v_wmma_f32_16x16x32_f16.
// Gate-interleaved weight packing lets the LSTM cell fuse into the GEMM
// epilogue per-lane (one kernel per timestep).
//   B=256, T=128, IN=128, H=1024, Z=256, 4H=4096, K_enc=K_dec=1152
// Block tiling: 2x4 (M-tile x N-group) rectangle -> WGP$ A+B reuse.
// The per-block B working set (~590KB/step) exceeds the WGP$, so demand loads
// would hit L2 right before each WMMA; branch-free global_prefetch two
// K-blocks ahead pulls weight lines WGP-local first. Loop bodies contain no
// data-dependent branches so the unroller can cluster loads across iterations.
// ---------------------------------------------------------------------------

typedef __attribute__((ext_vector_type(16))) _Float16 v16h;
typedef __attribute__((ext_vector_type(8)))  _Float16 v8h;
typedef __attribute__((ext_vector_type(8)))  float    v8f;

#define DEV __device__ __forceinline__

DEV v8f wmma_f16(v16h a, v16h b, v8f c) {
  return __builtin_amdgcn_wmma_f32_16x16x32_f16(false, a, false, b,
                                                (short)0, c, false, false);
}

// A fragment (16x32 f16): lane L holds row m0+(L&15); K = kb+(L>>4)*16 .. +15.
// One contiguous 32B load.
DEV v16h load_a_frag(const _Float16* __restrict__ A, int lda, int m0, int kb, int lane) {
  const int m  = m0 + (lane & 15);
  const int hi = lane >> 4;
  const _Float16* p = A + (size_t)m * lda + kb + hi * 16;
  return *(const v16h*)(p);
}

// B fragment (32x16 f16): lane L holds column n0+(L&15); K = kb+(L>>4)*16 .. +15.
// One contiguous 32B load from row-major W^T storage ([N][K]).
DEV v16h load_b_frag(const _Float16* __restrict__ Wt, int ldw, int n0, int kb, int lane) {
  const int n  = n0 + (lane & 15);
  const int hi = lane >> 4;
  return *(const v16h*)(Wt + (size_t)n * ldw + kb + hi * 16);
}

// Prefetch the cacheline a B fragment will need (one byte per lane; rows are
// distinct cachelines). No dest VGPR, no counter -> cannot be sunk by RA.
DEV void prefetch_b_frag(const _Float16* Wt, int ldw, int n0, int kb, int lane) {
  const int n  = n0 + (lane & 15);
  const int hi = lane >> 4;
  __builtin_prefetch(Wt + (size_t)n * ldw + kb + hi * 16, 0, 1);
}

DEV float sigf(float x) { return 1.0f / (1.0f + __expf(-x)); }

// ------------------------------- pack kernels ------------------------------

__global__ void f32_to_f16_kernel(const float* __restrict__ in,
                                  _Float16* __restrict__ out, int n) {
  int i = blockIdx.x * 256 + threadIdx.x;
  if (i < n) out[i] = (_Float16)in[i];
}

// Packed gate matrix G[4096][1152]: packed row p -> (group ng=p/64, gate=(p%64)/16,
// jj=p%16) -> original row r = gate*1024 + ng*16 + jj. Columns: k<128 from W_ih
// (row stride ldih), k>=128 from W_hh (row stride 1024).
__global__ void pack_gateW_kernel(const float* __restrict__ Wih, int ldih,
                                  const float* __restrict__ Whh,
                                  _Float16* __restrict__ G) {
  int idx = blockIdx.x * 256 + threadIdx.x;
  if (idx >= 4096 * 1152) return;
  int p = idx / 1152, k = idx - p * 1152;
  int ng = p >> 6, rem = p & 63, gate = rem >> 4, jj = rem & 15;
  int r = gate * 1024 + ng * 16 + jj;
  float v = (k < 128) ? Wih[(size_t)r * ldih + k] : Whh[(size_t)r * 1024 + (k - 128)];
  G[idx] = (_Float16)v;
}

__global__ void pack_gate_bias_kernel(const float* __restrict__ b1,
                                      const float* __restrict__ b2,
                                      float* __restrict__ out) {
  int p = blockIdx.x * 256 + threadIdx.x;
  if (p >= 4096) return;
  int ng = p >> 6, rem = p & 63, gate = rem >> 4, jj = rem & 15;
  int r = gate * 1024 + ng * 16 + jj;
  out[p] = b1[r] + b2[r];
}

// W_z slice of dec_W_ih (cols 128..383), packed rows, K=256.
__global__ void pack_Wz_kernel(const float* __restrict__ Wih,
                               _Float16* __restrict__ G) {
  int idx = blockIdx.x * 256 + threadIdx.x;
  if (idx >= 4096 * 256) return;
  int p = idx >> 8, k = idx & 255;
  int ng = p >> 6, rem = p & 63, gate = rem >> 4, jj = rem & 15;
  int r = gate * 1024 + ng * 16 + jj;
  G[idx] = (_Float16)Wih[(size_t)r * 384 + 128 + k];
}

// mean/log_var interleaved: group of 32 packed rows = 16 z-indices x {mean,lv}.
__global__ void pack_ml_kernel(const float* __restrict__ meanW,
                               const float* __restrict__ lvW,
                               const float* __restrict__ meanb,
                               const float* __restrict__ lvb,
                               _Float16* __restrict__ G,
                               float* __restrict__ bias) {
  int idx = blockIdx.x * 256 + threadIdx.x;
  if (idx >= 512 * 1024) return;
  int p = idx >> 10, k = idx & 1023;
  int zg = p >> 5, rem = p & 31, which = rem >> 4, jj = rem & 15;
  int r = zg * 16 + jj;
  G[idx] = (_Float16)(which ? lvW[(size_t)r * 1024 + k] : meanW[(size_t)r * 1024 + k]);
  if (k == 0) bias[p] = which ? lvb[r] : meanb[r];
}

// ----------------------- fused LSTM step (WMMA + cell) ---------------------
// gates[256][4096 packed] = [xpart | hprev] (K=128+1024) x G^T, then the full
// LSTM cell per lane (4 gate accumulators map to same (m, j)).
template<bool USE_GZ>
__global__ __launch_bounds__(256) void lstm_step_kernel(
    const _Float16* __restrict__ xpart, int xld,     // [256][128], row stride xld
    const _Float16* __restrict__ hprev,              // [256][1024]
    const _Float16* __restrict__ G,                  // [4096][1152] packed
    const float* __restrict__ bias_or_gz,            // bias[4096] or gz[256][4096]
    float* __restrict__ c_state,                     // [256][1024]
    _Float16* __restrict__ h_out) {                  // [256][1024]
  const int lane = threadIdx.x & 31;
  const int w    = threadIdx.x >> 5;                 // 0..7
  const int mtb  = blockIdx.x >> 4;                  // 0..7  (M block = 2 m-tiles)
  const int ngb  = blockIdx.x & 15;                  // 0..15 (N block = 4 ng groups)
  const int mt   = mtb * 2 + (w >> 2);               // 0..15
  const int ng   = ngb * 4 + (w & 3);                // 0..63
  const int m0 = mt * 16;
  const int p0 = ng * 64;

  v8f acc0 = {}, acc1 = {}, acc2 = {}, acc3 = {};

  // Prime the WGP-local cache for the first two K-blocks.
  prefetch_b_frag(G, 1152, p0 +  0, 0, lane);
  prefetch_b_frag(G, 1152, p0 + 16, 0, lane);
  prefetch_b_frag(G, 1152, p0 + 32, 0, lane);
  prefetch_b_frag(G, 1152, p0 + 48, 0, lane);
  prefetch_b_frag(G, 1152, p0 +  0, 32, lane);
  prefetch_b_frag(G, 1152, p0 + 16, 32, lane);
  prefetch_b_frag(G, 1152, p0 + 32, 32, lane);
  prefetch_b_frag(G, 1152, p0 + 48, 32, lane);

  // x contribution (K = 0..127); prefetch kg+64 is always a valid G column.
  for (int kg = 0; kg < 128; kg += 32) {
    prefetch_b_frag(G, 1152, p0 +  0, kg + 64, lane);
    prefetch_b_frag(G, 1152, p0 + 16, kg + 64, lane);
    prefetch_b_frag(G, 1152, p0 + 32, kg + 64, lane);
    prefetch_b_frag(G, 1152, p0 + 48, kg + 64, lane);
    v16h a = load_a_frag(xpart, xld, m0, kg, lane);
    acc0 = wmma_f16(a, load_b_frag(G, 1152, p0 +  0, kg, lane), acc0);
    acc1 = wmma_f16(a, load_b_frag(G, 1152, p0 + 16, kg, lane), acc1);
    acc2 = wmma_f16(a, load_b_frag(G, 1152, p0 + 32, kg, lane), acc2);
    acc3 = wmma_f16(a, load_b_frag(G, 1152, p0 + 48, kg, lane), acc3);
  }
  // h contribution (K = 128..1151); clamp is a select, not a branch.
  for (int kb = 0; kb < 1024; kb += 32) {
    const int kp = (kb + 64 < 1024) ? (128 + kb + 64) : 1120;
    prefetch_b_frag(G, 1152, p0 +  0, kp, lane);
    prefetch_b_frag(G, 1152, p0 + 16, kp, lane);
    prefetch_b_frag(G, 1152, p0 + 32, kp, lane);
    prefetch_b_frag(G, 1152, p0 + 48, kp, lane);
    v16h a = load_a_frag(hprev, 1024, m0, kb, lane);
    acc0 = wmma_f16(a, load_b_frag(G, 1152, p0 +  0, 128 + kb, lane), acc0);
    acc1 = wmma_f16(a, load_b_frag(G, 1152, p0 + 16, 128 + kb, lane), acc1);
    acc2 = wmma_f16(a, load_b_frag(G, 1152, p0 + 32, 128 + kb, lane), acc2);
    acc3 = wmma_f16(a, load_b_frag(G, 1152, p0 + 48, 128 + kb, lane), acc3);
  }

  const int nn = lane & 15, hi = lane >> 4;
  const int j = ng * 16 + nn;                          // hidden index
  float b0s = 0.f, b1s = 0.f, b2s = 0.f, b3s = 0.f;
  if (!USE_GZ) {
    b0s = bias_or_gz[p0 + nn];
    b1s = bias_or_gz[p0 + 16 + nn];
    b2s = bias_or_gz[p0 + 32 + nn];
    b3s = bias_or_gz[p0 + 48 + nn];
  }
#pragma unroll
  for (int v = 0; v < 8; ++v) {
    const int m = m0 + v + hi * 8;
    float gi, gf, gg, go;
    if (USE_GZ) {
      const float* gzr = bias_or_gz + (size_t)m * 4096 + p0 + nn;
      gi = acc0[v] + gzr[0];
      gf = acc1[v] + gzr[16];
      gg = acc2[v] + gzr[32];
      go = acc3[v] + gzr[48];
    } else {
      gi = acc0[v] + b0s; gf = acc1[v] + b1s; gg = acc2[v] + b2s; go = acc3[v] + b3s;
    }
    const float i_ = sigf(gi), f_ = sigf(gf), g_ = tanhf(gg), o_ = sigf(go);
    const size_t idx = (size_t)m * 1024 + j;
    const float cn = f_ * c_state[idx] + i_ * g_;
    c_state[idx] = cn;
    h_out[idx] = (_Float16)(o_ * tanhf(cn));
  }
}

// gz[256][4096 packed] = z(f16,[256][256]) x Wz^T + (b_ih+b_hh), WMMA.
__global__ __launch_bounds__(256) void gz_init_kernel(
    const _Float16* __restrict__ z, const _Float16* __restrict__ G,
    const float* __restrict__ bias, float* __restrict__ gz) {
  const int lane = threadIdx.x & 31;
  const int wave = blockIdx.x * 8 + (threadIdx.x >> 5);
  const int mt = wave >> 6, ng = wave & 63;
  const int m0 = mt * 16, p0 = ng * 64;
  v8f acc0 = {}, acc1 = {}, acc2 = {}, acc3 = {};
  for (int kb = 0; kb < 256; kb += 32) {
    v16h a  = load_a_frag(z, 256, m0, kb, lane);
    acc0 = wmma_f16(a, load_b_frag(G, 256, p0 +  0, kb, lane), acc0);
    acc1 = wmma_f16(a, load_b_frag(G, 256, p0 + 16, kb, lane), acc1);
    acc2 = wmma_f16(a, load_b_frag(G, 256, p0 + 32, kb, lane), acc2);
    acc3 = wmma_f16(a, load_b_frag(G, 256, p0 + 48, kb, lane), acc3);
  }
  const int nn = lane & 15, hi = lane >> 4;
#pragma unroll
  for (int v = 0; v < 8; ++v) {
    const int m = m0 + v + hi * 8;
    float* gzr = gz + (size_t)m * 4096 + p0 + nn;
    gzr[0]  = acc0[v] + bias[p0 + nn];
    gzr[16] = acc1[v] + bias[p0 + 16 + nn];
    gzr[32] = acc2[v] + bias[p0 + 32 + nn];
    gzr[48] = acc3[v] + bias[p0 + 48 + nn];
  }
}

// mean/lv GEMM with fused reparameterization: z = noise*exp(0.5*lv) + mean.
__global__ __launch_bounds__(256) void mean_lv_sample_kernel(
    const _Float16* __restrict__ hT, const _Float16* __restrict__ G,
    const float* __restrict__ bias, const float* __restrict__ noise,
    _Float16* __restrict__ z_out) {
  const int lane = threadIdx.x & 31;
  const int wave = blockIdx.x * 8 + (threadIdx.x >> 5);   // 0..255
  const int mt = wave >> 4, zg = wave & 15;
  const int m0 = mt * 16;
  v8f acc0 = {}, acc1 = {};
  for (int kb = 0; kb < 1024; kb += 32) {
    v16h a = load_a_frag(hT, 1024, m0, kb, lane);
    acc0 = wmma_f16(a, load_b_frag(G, 1024, zg * 32 +  0, kb, lane), acc0);
    acc1 = wmma_f16(a, load_b_frag(G, 1024, zg * 32 + 16, kb, lane), acc1);
  }
  const int nn = lane & 15, hi = lane >> 4;
  const int jz = zg * 16 + nn;
  const float bm = bias[zg * 32 + nn], bl = bias[zg * 32 + 16 + nn];
#pragma unroll
  for (int v = 0; v < 8; ++v) {
    const int m = m0 + v + hi * 8;
    const float mean = acc0[v] + bm;
    const float lv   = acc1[v] + bl;
    const float zz = noise[(size_t)m * 256 + jz] * __expf(0.5f * lv) + mean;
    z_out[(size_t)m * 256 + jz] = (_Float16)zz;
  }
}

// Generic small projection: C[256][Ntiles*16] = A(f16) x W^T + bias; optional
// f32 and/or f16 outputs. Used for init-decoder and output projection.
__global__ __launch_bounds__(256) void proj_kernel(
    const _Float16* __restrict__ A, int lda,
    const _Float16* __restrict__ W, int ldw,     // [N][K]
    const float* __restrict__ bias,
    float* __restrict__ out32, int out32_stride,
    _Float16* __restrict__ out16, int out16_stride,
    int Ntiles, int K) {
  const int lane = threadIdx.x & 31;
  const int wave = blockIdx.x * 8 + (threadIdx.x >> 5);
  const int mt = wave / Ntiles, nt = wave - mt * Ntiles;
  const int m0 = mt * 16;
  v8f acc = {};
  for (int kb = 0; kb < K; kb += 32) {
    const int kp = (kb + 64 < K) ? kb + 64 : kb;
    prefetch_b_frag(W, ldw, nt * 16, kp, lane);
    v16h a = load_a_frag(A, lda, m0, kb, lane);
    v16h b = load_b_frag(W, ldw, nt * 16, kb, lane);
    acc = wmma_f16(a, b, acc);
  }
  const int nn = lane & 15, hi = lane >> 4;
  const int n = nt * 16 + nn;
  const float bv = bias[n];
#pragma unroll
  for (int v = 0; v < 8; ++v) {
    const int m = m0 + v + hi * 8;
    const float val = acc[v] + bv;
    if (out32) out32[(size_t)m * out32_stride + n] = val;
    if (out16) out16[(size_t)m * out16_stride + n] = (_Float16)val;
  }
}

// --------------------------------- launch ----------------------------------

extern "C" void kernel_launch(void* const* d_in, const int* in_sizes, int n_in,
                              void* d_out, int out_size, void* d_ws, size_t ws_size,
                              hipStream_t stream) {
  const float* x         = (const float*)d_in[0];
  const float* noise     = (const float*)d_in[1];
  const float* enc_W_ih  = (const float*)d_in[2];
  const float* enc_W_hh  = (const float*)d_in[3];
  const float* enc_b_ih  = (const float*)d_in[4];
  const float* enc_b_hh  = (const float*)d_in[5];
  const float* mean_W    = (const float*)d_in[6];
  const float* mean_b    = (const float*)d_in[7];
  const float* lv_W      = (const float*)d_in[8];
  const float* lv_b      = (const float*)d_in[9];
  const float* initdec_W = (const float*)d_in[10];
  const float* initdec_b = (const float*)d_in[11];
  const float* dec_W_ih  = (const float*)d_in[12];
  const float* dec_W_hh  = (const float*)d_in[13];
  const float* dec_b_ih  = (const float*)d_in[14];
  const float* dec_b_hh  = (const float*)d_in[15];
  const float* out_W     = (const float*)d_in[16];
  const float* out_b     = (const float*)d_in[17];
  float* out             = (float*)d_out;

  char* ws = (char*)d_ws;
  size_t off = 0;
  auto carve = [&](size_t bytes) -> char* {
    char* p = ws + off;
    off += (bytes + 255) & ~(size_t)255;
    return p;
  };
  _Float16* x16     = (_Float16*)carve((size_t)256 * 128 * 128 * 2); // 8 MB
  _Float16* encG    = (_Float16*)carve((size_t)4096 * 1152 * 2);     // 9 MB
  _Float16* decG    = (_Float16*)carve((size_t)4096 * 1152 * 2);     // 9 MB
  _Float16* WzG     = (_Float16*)carve((size_t)4096 * 256 * 2);      // 2 MB
  _Float16* mlG     = (_Float16*)carve((size_t)512 * 1024 * 2);      // 1 MB
  _Float16* idW16   = (_Float16*)carve((size_t)128 * 256 * 2);
  _Float16* outW16  = (_Float16*)carve((size_t)128 * 1024 * 2);
  float*    encBias = (float*)carve(4096 * 4);
  float*    decBias = (float*)carve(4096 * 4);
  float*    mlBias  = (float*)carve(512 * 4);
  _Float16* hA      = (_Float16*)carve((size_t)256 * 1024 * 2);
  _Float16* hB      = (_Float16*)carve((size_t)256 * 1024 * 2);
  float*    cst     = (float*)carve((size_t)256 * 1024 * 4);
  _Float16* z16     = (_Float16*)carve((size_t)256 * 256 * 2);
  float*    gz      = (float*)carve((size_t)256 * 4096 * 4);         // 4 MB
  _Float16* xin16   = (_Float16*)carve((size_t)256 * 128 * 2);
  (void)ws_size; (void)in_sizes; (void)n_in; (void)out_size;

  // ---- weight / input packing (f32 -> f16, gate-interleaved layouts) ----
  f32_to_f16_kernel<<<(256 * 128 * 128 + 255) / 256, 256, 0, stream>>>(x, x16, 256 * 128 * 128);
  pack_gateW_kernel<<<(4096 * 1152 + 255) / 256, 256, 0, stream>>>(enc_W_ih, 128, enc_W_hh, encG);
  pack_gate_bias_kernel<<<16, 256, 0, stream>>>(enc_b_ih, enc_b_hh, encBias);
  pack_gateW_kernel<<<(4096 * 1152 + 255) / 256, 256, 0, stream>>>(dec_W_ih, 384, dec_W_hh, decG);
  pack_gate_bias_kernel<<<16, 256, 0, stream>>>(dec_b_ih, dec_b_hh, decBias);
  pack_Wz_kernel<<<(4096 * 256 + 255) / 256, 256, 0, stream>>>(dec_W_ih, WzG);
  pack_ml_kernel<<<(512 * 1024 + 255) / 256, 256, 0, stream>>>(mean_W, lv_W, mean_b, lv_b, mlG, mlBias);
  f32_to_f16_kernel<<<(128 * 256 + 255) / 256, 256, 0, stream>>>(initdec_W, idW16, 128 * 256);
  f32_to_f16_kernel<<<(128 * 1024 + 255) / 256, 256, 0, stream>>>(out_W, outW16, 128 * 1024);

  // ---- encoder scan ----
  (void)hipMemsetAsync(hA, 0, (size_t)256 * 1024 * 2, stream);
  (void)hipMemsetAsync(cst, 0, (size_t)256 * 1024 * 4, stream);
  const _Float16* hc = hA;
  _Float16* hn = hB;
  for (int t = 0; t < 128; ++t) {
    lstm_step_kernel<false><<<128, 256, 0, stream>>>(
        x16 + (size_t)t * 128, 128 * 128, hc, encG, encBias, cst, hn);
    const _Float16* tmp = hc; hc = hn; hn = (_Float16*)tmp;
  }
  // hc now holds hT

  // ---- reparameterize + decoder init ----
  mean_lv_sample_kernel<<<32, 256, 0, stream>>>(hc, mlG, mlBias, noise, z16);
  gz_init_kernel<<<128, 256, 0, stream>>>(z16, WzG, decBias, gz);
  proj_kernel<<<16, 256, 0, stream>>>(z16, 256, idW16, 256, initdec_b,
                                      nullptr, 0, xin16, 128, 8, 256);

  // ---- decoder scan ----
  (void)hipMemsetAsync(hA, 0, (size_t)256 * 1024 * 2, stream);
  (void)hipMemsetAsync(cst, 0, (size_t)256 * 1024 * 4, stream);
  hc = hA; hn = hB;
  for (int t = 0; t < 128; ++t) {
    lstm_step_kernel<true><<<128, 256, 0, stream>>>(
        xin16, 128, hc, decG, gz, cst, hn);
    // out_t = h_t @ out_W^T + out_b -> d_out[:, t, :] (f32) and next x_in (f16)
    proj_kernel<<<16, 256, 0, stream>>>(hn, 1024, outW16, 1024, out_b,
                                        out + (size_t)t * 128, 128 * 128,
                                        xin16, 128, 8, 1024);
    const _Float16* tmp = hc; hc = hn; hn = (_Float16*)tmp;
  }
}
